// StateEncoder_11201274708331
// MI455X (gfx1250) — matrix operand order repaired and checked by reading, not compiled, not run on hardware
//
#include <hip/hip_runtime.h>
#include <hip/hip_bf16.h>
#include <stdint.h>

// ---------------- problem constants ----------------
#define BB   4
#define LL   128
#define DG   512
#define DSP  256
#define HH   512
#define OUTC 640

// fused-kernel LDS geometry (ushort units unless noted)
#define PSTR       40            // per-wave A patch row stride (80B: conflict-free b128)
#define PATCH_USH  (16 * PSTR)   // 640 ushorts per wave
#define BSTR       520           // B column stride (1040B: conflict-free b128), made by TDM pad
#define BUF_USH    (16 * BSTR)   // 8320 ushorts per B buffer (16 cols x 512 + pad)
#define BOFF_USH   (8 * PATCH_USH)               // B buffers after 8 wave patches
#define LDS_TOTAL  (BOFF_USH + 2 * BUF_USH)      // 21760 ushorts = 43520 B

typedef __attribute__((ext_vector_type(16))) __bf16 v16bf;
typedef __attribute__((ext_vector_type(8)))  float  v8f;
typedef unsigned int v4u __attribute__((ext_vector_type(4)));
typedef int          v4i __attribute__((ext_vector_type(4)));
typedef int          v8i __attribute__((ext_vector_type(8)));

__device__ __forceinline__ unsigned short f2bf(float f) {
    uint32_t u = __float_as_uint(f);
    u += 0x7FFFu + ((u >> 16) & 1u);
    return (unsigned short)(u >> 16);
}

union FragU { uint4 q[2]; v16bf v; };

// ---------------- TDM: load one 16-col B tile (16KB contiguous) into LDS ----------------
// 2D tile: dim0 = 512 bf16 (one column of W2t), dim1 = 16 columns. pad_enable inserts
// 4 DWORDs (8 bf16) after every 256 DWORDs (512 bf16) -> LDS column stride = 520 elems.
__device__ __forceinline__ void tdm_load_btile(const unsigned short* W2t, int nt,
                                               uint32_t lds_byte_addr) {
    const uint64_t gaddr = (uint64_t)(uintptr_t)W2t + (uint64_t)nt * (16u * HH * 2u);
    v4u g0;
    g0.x = 1u;                                           // count=1, user mode, no gather
    g0.y = lds_byte_addr;                                // [63:32] lds_addr
    g0.z = (uint32_t)gaddr;                              // [95:64] global_addr lo
    g0.w = (uint32_t)((gaddr >> 32) & 0x01FFFFFFull)     // [120:96] global_addr hi
         | (2u << 30);                                   // [127:126] type = 2 (image)
    v8i g1;
    g1[0] = (int)((1u << 16)      // data_size = 1 (2 bytes)
                | (1u << 20)      // pad_enable
                | (7u << 22)      // pad_interval = 256 DWORDs
                | (3u << 25));    // pad_amount  = 4 DWORDs
    g1[1] = (int)(512u << 16);    // tensor_dim0 = 512 (low 16 in [63:48])
    g1[2] = (int)(16u  << 16);    // tensor_dim0 hi = 0; tensor_dim1 = 16 (low 16 in [95:80])
    g1[3] = (int)(512u << 16);    // tensor_dim1 hi = 0; tile_dim0 = 512 ([127:112])
    g1[4] = 16;                   // tile_dim1 = 16 ([143:128]); tile_dim2 = 0
    g1[5] = 512;                  // tensor_dim0_stride = 512 ([207:160] low 32)
    g1[6] = 0;
    g1[7] = 0;
    const v4i z4 = (v4i)0;
#if __has_include(<hip/amd_detail/amd_gfx1250_TDM.h>)
    const v8i z8 = (v8i)0;        // therock/clang-23 toolchain: 6-arg builtin
    __builtin_amdgcn_tensor_load_to_lds(g0, g1, z4, z4, z8, 0);
#else                             // ROCm 7.2 / clang-22: 5-arg builtin
    __builtin_amdgcn_tensor_load_to_lds(g0, g1, z4, z4, 0);
#endif
}

// ---------------- kernel 1: projections ----------------
__global__ __launch_bounds__(256) void se_proj(
    const float* __restrict__ go, const float* __restrict__ sp,
    const float* __restrict__ W_utt, const float* __restrict__ b_utt,
    const float* __restrict__ W_sp,  const float* __restrict__ b_sp,
    float* __restrict__ gj, float* __restrict__ gi,
    float* __restrict__ sj, float* __restrict__ si)
{
    __shared__ float xg[DG];
    __shared__ float xs[DSP];
    const int bl = blockIdx.x;                 // 0 .. B*L-1
    const float* grow = go + (size_t)bl * DG;
    const float* srow = sp + (size_t)bl * DSP;

    for (int d = threadIdx.x; d < DG; d += 256)  xg[d] = grow[d];
    if (threadIdx.x < DSP)                       xs[threadIdx.x] = srow[threadIdx.x];
    __syncthreads();

    for (int h = threadIdx.x; h < HH; h += 256) {
        float aj = 0.f, ai = 0.f;
        #pragma unroll 4
        for (int d = 0; d < DG; ++d) {
            float x = xg[d];
            aj = fmaf(x, W_utt[(size_t)d * HH + h],        aj);
            ai = fmaf(x, W_utt[(size_t)(DG + d) * HH + h], ai);
        }
        float cj = 0.f, ci = 0.f;
        #pragma unroll 4
        for (int d = 0; d < DSP; ++d) {
            float x = xs[d];
            cj = fmaf(x, W_sp[(size_t)d * HH + h],         cj);
            ci = fmaf(x, W_sp[(size_t)(DSP + d) * HH + h], ci);
        }
        size_t o = (size_t)bl * HH + h;
        gj[o] = aj;
        gi[o] = ai + b_utt[h];
        sj[o] = cj;
        si[o] = ci + b_sp[h];
    }
}

// ---------------- kernel 2: W2 transpose + bf16 convert ----------------
__global__ __launch_bounds__(256) void se_w2t(
    const float* __restrict__ W2, unsigned short* __restrict__ W2t)
{
    int idx = blockIdx.x * 256 + threadIdx.x;
    if (idx < OUTC * HH) {
        int o = idx >> 9;
        int h = idx & (HH - 1);
        W2t[idx] = f2bf(W2[(size_t)h * OUTC + o]);
    }
}

// ---------------- kernel 3: fused pairwise tanh/mix + WMMA GEMM ----------------
// One block per (b,i): 128 output rows (all j). Wave w owns rows j = 16w..16w+15.
// B tiles (16 cols of W2t^T) are TDM-streamed into a double-buffered LDS region and
// shared by all 8 waves; A fragments live in registers.
__global__ __launch_bounds__(256) void se_fused(
    const float* __restrict__ gj, const float* __restrict__ gi,
    const float* __restrict__ sj, const float* __restrict__ si,
    const unsigned short* __restrict__ W2t,
    const float* __restrict__ b2,
    const float* __restrict__ mix_w, const float* __restrict__ gammap,
    float* __restrict__ out)
{
    __shared__ __align__(16) unsigned short lds_all[LDS_TOTAL];

    const int b = blockIdx.x >> 7;
    const int i = blockIdx.x & (LL - 1);

    const int wave  = threadIdx.x >> 5;
    const int lane  = threadIdx.x & 31;
    const int lrow  = lane & 15;
    const int khalf = lane >> 4;

    const uint32_t lds_base = (uint32_t)(uintptr_t)(&lds_all[0]);

    // Kick off TDM load of B tile 0 immediately; it overlaps the tanh-heavy A phase.
    if (wave == 0)
        tdm_load_btile(W2t, 0, lds_base + 2u * BOFF_USH);

    // ScalarMix: gamma * softmax(mix_w)
    const float m0 = mix_w[0], m1 = mix_w[1];
    const float mx = fmaxf(m0, m1);
    const float e0 = __expf(m0 - mx), e1 = __expf(m1 - mx);
    const float g  = gammap[0] / (e0 + e1);
    const float w0 = e0 * g, w1 = e1 * g;

    const float* gj_b   = gj + (size_t)(b * LL) * HH;
    const float* sj_b   = sj + (size_t)(b * LL) * HH;
    const float* gi_row = gi + (size_t)(b * LL + i) * HH;
    const float* si_row = si + (size_t)(b * LL + i) * HH;

    // Phase 1 (per wave, no cross-wave sync): build 16x512 bf16 A tile 32 k at a time
    // through a private LDS patch, pulling each WMMA A fragment into registers.
    unsigned short* patch = &lds_all[wave * PATCH_USH];
    const int j0w = wave * 16;
    FragU afrag[16];
    #pragma unroll
    for (int kc = 0; kc < 16; ++kc) {
        const int hb = kc * 32;
        const float giv = gi_row[hb + lane];
        const float siv = si_row[hb + lane];
        #pragma unroll
        for (int r = 0; r < 16; ++r) {
            const float u = tanhf(gj_b[(size_t)(j0w + r) * HH + hb + lane] + giv);
            const float s = tanhf(sj_b[(size_t)(j0w + r) * HH + hb + lane] + siv);
            patch[r * PSTR + lane] = f2bf(w0 * u + w1 * s);   // row r, k = hb+lane
        }
        // same-wave DS ops are in-order: fragment read sees the stores above.
        const int abase = lrow * PSTR + khalf * 8;
        afrag[kc].q[0] = *reinterpret_cast<const uint4*>(&patch[abase]);
        afrag[kc].q[1] = *reinterpret_cast<const uint4*>(&patch[abase + 16]);
    }

    const size_t rowbase = (size_t)blockIdx.x * LL + (size_t)j0w;

    // Phase 2: all 8 waves sweep the 40 N-tiles together; TDM double-buffers B.
    for (int nt = 0; nt < OUTC / 16; ++nt) {
        if (wave == 0)
            __builtin_amdgcn_s_wait_tensorcnt(0);   // B tile `nt` resident in LDS
        __syncthreads();                            // visible to all waves
        if (wave == 0 && nt + 1 < OUTC / 16)        // prefetch next tile (other buffer)
            tdm_load_btile(W2t, nt + 1,
                           lds_base + 2u * (BOFF_USH + ((nt + 1) & 1) * BUF_USH));

        const unsigned short* bcol =
            &lds_all[BOFF_USH + (nt & 1) * BUF_USH] + lrow * BSTR + khalf * 8;

        v8f acc = {};
        #pragma unroll
        for (int k0 = 0; k0 < 16; ++k0) {
            FragU fb;
            fb.q[0] = *reinterpret_cast<const uint4*>(bcol + k0 * 32);
            fb.q[1] = *reinterpret_cast<const uint4*>(bcol + k0 * 32 + 16);
            acc = __builtin_amdgcn_wmma_f32_16x16x32_bf16(
                      false, afrag[k0].v, false, fb.v, (short)0, acc, false, false);
        }

        // Epilogue: C/D f32 layout — lane -> N col, VGPR r -> M row r + 8*khalf
        const int col = nt * 16 + lrow;
        const float bias = b2[col];
        #pragma unroll
        for (int r = 0; r < 8; ++r) {
            const int m = r + khalf * 8;
            out[(rowbase + m) * OUTC + col] = tanhf(acc[r] + bias);
        }
    }
}

// ---------------- launcher ----------------
extern "C" void kernel_launch(void* const* d_in, const int* in_sizes, int n_in,
                              void* d_out, int out_size, void* d_ws, size_t ws_size,
                              hipStream_t stream) {
    const float* go    = (const float*)d_in[0];
    const float* sp    = (const float*)d_in[1];
    const float* W_utt = (const float*)d_in[2];
    const float* b_utt = (const float*)d_in[3];
    const float* W_sp  = (const float*)d_in[4];
    const float* b_sp  = (const float*)d_in[5];
    const float* W2    = (const float*)d_in[6];
    const float* b2    = (const float*)d_in[7];
    const float* mixw  = (const float*)d_in[8];
    const float* gam   = (const float*)d_in[9];
    float* out = (float*)d_out;

    const size_t NPROJ = (size_t)BB * LL * HH;
    float* gj = (float*)d_ws;
    float* gi = gj + NPROJ;
    float* sj = gi + NPROJ;
    float* si = sj + NPROJ;
    unsigned short* W2t = (unsigned short*)(si + NPROJ);

    se_proj<<<BB * LL, 256, 0, stream>>>(go, sp, W_utt, b_utt, W_sp, b_sp,
                                         gj, gi, sj, si);
    se_w2t<<<(OUTC * HH + 255) / 256, 256, 0, stream>>>(W2, W2t);
    se_fused<<<BB * LL, 256, 0, stream>>>(gj, gi, sj, si, W2t, b2, mixw, gam, out);
}
